// HybridSeq2Seq_22634477650598
// MI455X (gfx1250) — compile-verified, matrix operand order checked by reading, not run on hardware
//
#include <hip/hip_runtime.h>
#include <math.h>

typedef __attribute__((ext_vector_type(16))) __bf16 v16bf;
typedef __attribute__((ext_vector_type(8)))  float  v8f;

#define DEV static __device__ __forceinline__

namespace {

constexpr int Vv = 32000;
constexpr int E  = 256;
constexpr int H  = 384;
constexpr int Bn = 32;
constexpr int S  = 256;
constexpr int T  = 60;
constexpr int HD = 768;        // 2*H
constexpr int G1 = 4 * H;      // 1536
constexpr int GD = 4 * HD;     // 3072
constexpr int HP = 400;        // padded LDS row for h (bf16)

union FragU { uint4 u4[2]; v16bf v; };

// A/B fragment load for v_wmma_f32_16x16x32_bf16.
// ISA 7.12.2 16-bit A 16x32 layout: lanes0-15 row M=lane, K={0..7,16..23};
// lanes16-31 row M=lane-16, K={8..15,24..31}. B (N x K, pre-transposed
// weights) uses the mirrored layout.
DEV v16bf load_frag(const __bf16* base, int ld, int row0, int k0, int rmax) {
  int lane = threadIdx.x & 31;
  int half = lane >> 4;
  int r = row0 + (lane & 15);
  if (r > rmax) r = rmax;                    // clamp: OOB rows masked at store
  const __bf16* p = base + (size_t)r * ld + k0 + half * 8;
  FragU f;
  f.u4[0] = *reinterpret_cast<const uint4*>(p);       // K = 8*half + 0..7
  f.u4[1] = *reinterpret_cast<const uint4*>(p + 16);  // K = 16 + 8*half + 0..7
  return f.v;
}

DEV v8f wmma_bf16(v16bf a, v16bf b, v8f c) {
  return __builtin_amdgcn_wmma_f32_16x16x32_bf16(
      false, a, false, b, (short)0, c, false, false);
}

// C/D 16x16 f32 layout: VGPR r -> (M=r,N=lane) lanes0-15, (M=8+r,N=lane-16).
DEV void store_tile(float* C, size_t ldc, int m0, int n0, int M,
                    v8f acc, const float* bias) {
  int lane = threadIdx.x & 31;
  int col = n0 + (lane & 15);
  int rb = m0 + (lane >> 4) * 8;
  float bv = bias ? bias[col] : 0.f;
#pragma unroll
  for (int r = 0; r < 8; ++r) {
    int row = rb + r;
    if (row < M) C[(size_t)row * ldc + col] = acc[r] + bv;
  }
}

DEV float sigm(float x) { return 1.f / (1.f + __expf(-x)); }

// ---------------------------------------------------------------- utilities
__global__ void cast_f32_bf16(const float* __restrict__ s,
                              __bf16* __restrict__ d, size_t n) {
  size_t i = (size_t)blockIdx.x * blockDim.x + threadIdx.x;
  size_t st = (size_t)gridDim.x * blockDim.x;
  for (; i < n; i += st) d[i] = (__bf16)s[i];
}

__global__ void gather_src(const int* __restrict__ src,
                           const float* __restrict__ emb,
                           __bf16* __restrict__ xbf) {
  int row = blockIdx.x;                 // b*S + s, 0..8191
  int e   = threadIdx.x;                // 0..255
  int id  = src[row];
  xbf[(size_t)row * E + e] = (__bf16)emb[(size_t)id * E + e];
}

// ------------------------------------------------------------ generic GEMM
// C(MxN,f32) = A(MxK,bf16,row) * Wt(NxK,bf16,row)^T + bias.
// 8 waves arranged WM x WN (WM*WN == 8); wave tile = 16M x 32N (2 wmma),
// block tile = (WM*16)M x (WN*32)N, K stepped by 32.
//   <4,2>: 64x64 tile, for tall encoder GEMMs (M = 8192)
//   <2,4>: 32x128 tile, for the decoder FC (M = 32) -> no wasted waves
template <int WM, int WN>
__global__ __launch_bounds__(256) void gemm_bf16(
    const __bf16* __restrict__ A, int lda,
    const __bf16* __restrict__ Wt, int ldw,
    const float* __restrict__ bias,
    float* __restrict__ C, size_t ldc,
    int M, int N, int K) {
  int wid = threadIdx.x >> 5;
  int m0 = blockIdx.y * (WM * 16) + (wid % WM) * 16;
  int n0 = blockIdx.x * (WN * 32) + (wid / WM) * 32;
  v8f acc0 = {}, acc1 = {};
  for (int k0 = 0; k0 < K; k0 += 32) {
    // stream weights ahead into cache (global_prefetch_b8)
    __builtin_prefetch((const void*)(Wt + (size_t)(n0 + (threadIdx.x & 31)) * ldw
                                     + k0 + 128), 0, 1);
    v16bf a  = load_frag(A,  lda, m0,      k0, M - 1);
    v16bf b0 = load_frag(Wt, ldw, n0,      k0, N - 1);
    v16bf b1 = load_frag(Wt, ldw, n0 + 16, k0, N - 1);
    acc0 = wmma_bf16(a, b0, acc0);
    acc1 = wmma_bf16(a, b1, acc1);
  }
  store_tile(C, ldc, m0, n0,      M, acc0, bias);
  store_tile(C, ldc, m0, n0 + 16, M, acc1, bias);
}

// --------------------------------------------------- persistent LSTM scan
// gridDim.x == 2 (block0=fwd, block1=bwd), 768 threads = 24 waves.
// Wave w owns gate columns [16w,16w+16) of each of the 4 gates (8 wmma tiles:
// 4 gates x 2 M-tiles); h(bf16),c(f32) live in LDS across all S steps.
// NOTE: the K loop must NOT be unrolled -- fully unrolling makes all 48
// Whh B-fragments (384 VGPRs) loop-invariant w.r.t. `step`, and the
// compiler hoists + spills them to scratch (seen in round-2 disasm).
// With unroll disabled the per-iteration live set is ~112 VGPRs and the
// weights are re-read from WGP$/L2 (Whh is only 1.18 MB, cache-hot).
__global__ __launch_bounds__(768) void lstm_scan(
    const float* __restrict__ xg_f, const float* __restrict__ xg_b,
    const __bf16* __restrict__ Whh_f, const __bf16* __restrict__ Whh_b,
    __bf16* __restrict__ out_bf, int out_ld,
    float* __restrict__ out_f32,          // nullable (layer2: enc_out f32)
    float* __restrict__ hfin,             // nullable [32 x 768], dir*384 off
    float* __restrict__ cfin) {
  __shared__ __align__(16) __bf16 h_s[Bn * HP];
  __shared__ __align__(16) float  c_s[Bn * H];
  int dir = blockIdx.x;
  const float*  xg  = dir ? xg_b  : xg_f;
  const __bf16* Whh = dir ? Whh_b : Whh_f;

  for (int i = threadIdx.x; i < Bn * HP; i += blockDim.x) h_s[i] = (__bf16)0.f;
  for (int i = threadIdx.x; i < Bn * H;  i += blockDim.x) c_s[i] = 0.f;
  __syncthreads();

  int wid  = threadIdx.x >> 5;   // 0..23
  int lane = threadIdx.x & 31;
  int n0   = wid * 16;
  int col  = n0 + (lane & 15);

  for (int step = 0; step < S; ++step) {
    int s = dir ? (S - 1 - step) : step;
    v8f acc[4][2] = {};
#pragma clang loop unroll(disable)
    for (int k0 = 0; k0 < H; k0 += 32) {
      v16bf a0 = load_frag(h_s, HP, 0,  k0, Bn - 1);
      v16bf a1 = load_frag(h_s, HP, 16, k0, Bn - 1);
#pragma unroll
      for (int g = 0; g < 4; ++g) {
        v16bf bfr = load_frag(Whh, H, g * H + n0, k0, G1 - 1);
        acc[g][0] = wmma_bf16(a0, bfr, acc[g][0]);
        acc[g][1] = wmma_bf16(a1, bfr, acc[g][1]);
      }
    }
    __syncthreads();   // all reads of h_s done before this step's writes
#pragma unroll
    for (int mt = 0; mt < 2; ++mt) {
      int rb = mt * 16 + (lane >> 4) * 8;
#pragma unroll
      for (int r = 0; r < 8; ++r) {
        int b = rb + r;                            // batch row 0..31
        size_t xrow = ((size_t)b * S + s) * (size_t)G1;
        float gi = acc[0][mt][r] + xg[xrow + 0 * H + col];
        float gf = acc[1][mt][r] + xg[xrow + 1 * H + col];
        float gg = acc[2][mt][r] + xg[xrow + 2 * H + col];
        float go = acc[3][mt][r] + xg[xrow + 3 * H + col];
        float cn = sigm(gf) * c_s[b * H + col] + sigm(gi) * tanhf(gg);
        float hn = sigm(go) * tanhf(cn);
        c_s[b * H + col] = cn;
        h_s[b * HP + col] = (__bf16)hn;
        size_t orow = ((size_t)b * S + s) * (size_t)out_ld + dir * H + col;
        out_bf[orow] = (__bf16)hn;
        if (out_f32) out_f32[orow] = hn;
      }
    }
    __syncthreads();   // h_s/c_s updated before next step's reads
  }
  if (hfin) {
    for (int i = threadIdx.x; i < Bn * H; i += blockDim.x) {
      int b = i / H, n = i % H;
      hfin[b * HD + dir * H + n] = (float)h_s[b * HP + n];
      cfin[b * HD + dir * H + n] = c_s[b * H + n];
    }
  }
}

// -------------------------------------------------- decoder attention step
// one block per batch row; fused q = h@Wh^T, energy+softmax over S, ctx,
// and the concat writes (inp = [e_t, ctx], hctx[768:] = ctx).
__global__ __launch_bounds__(256) void attn_step(
    const float* __restrict__ h,          // 32 x 768 (current)
    const float* __restrict__ enc_part,   // 8192 x 384 (incl. attn_b)
    const float* __restrict__ enc_out,    // 8192 x 768 f32
    const float* __restrict__ attn_Wh,    // 384 x 768 f32
    const float* __restrict__ vvec,       // 384
    const int*   __restrict__ trg,        // B x T
    const float* __restrict__ emb,        // V x E
    __bf16* __restrict__ inp,             // 32 x 1024
    __bf16* __restrict__ hctx,            // 32 x 1536 (write [768:1536))
    int t) {
  __shared__ float hb_s[HD];
  __shared__ float q_s[H];
  __shared__ float sc_s[S];
  __shared__ float red_s[256];
  int b = blockIdx.x, tid = threadIdx.x;

  for (int i = tid; i < HD; i += 256) hb_s[i] = h[b * HD + i];
  __syncthreads();
  for (int hh = tid; hh < H; hh += 256) {
    const float* w = attn_Wh + (size_t)hh * HD;
    float a = 0.f;
    for (int k = 0; k < HD; ++k) a += hb_s[k] * w[k];
    q_s[hh] = a;
  }
  __syncthreads();
  {
    int s = tid;
    const float* ep = enc_part + ((size_t)b * S + s) * (size_t)H;
    float e = 0.f;
    for (int k = 0; k < H; ++k) e += tanhf(q_s[k] + ep[k]) * vvec[k];
    sc_s[s] = e;
  }
  __syncthreads();
  red_s[tid] = sc_s[tid];
  __syncthreads();
  for (int o = 128; o > 0; o >>= 1) {
    if (tid < o) red_s[tid] = fmaxf(red_s[tid], red_s[tid + o]);
    __syncthreads();
  }
  float mx = red_s[0];
  __syncthreads();
  float ex = __expf(sc_s[tid] - mx);
  red_s[tid] = ex;
  __syncthreads();
  for (int o = 128; o > 0; o >>= 1) {
    if (tid < o) red_s[tid] += red_s[tid + o];
    __syncthreads();
  }
  float inv = 1.f / red_s[0];
  __syncthreads();
  sc_s[tid] = ex * inv;
  __syncthreads();
  for (int d = tid; d < HD; d += 256) {
    float a = 0.f;
    for (int s = 0; s < S; ++s)
      a += sc_s[s] * enc_out[((size_t)b * S + s) * (size_t)HD + d];
    inp[(size_t)b * (E + HD) + E + d] = (__bf16)a;
    hctx[(size_t)b * (2 * HD) + HD + d] = (__bf16)a;
  }
  int id = trg[b * T + t];
  if (tid < E) inp[(size_t)b * (E + HD) + tid] = (__bf16)emb[(size_t)id * E + tid];
}

// --------------------------------------------- decoder gate GEMM + cell
// 48 blocks x 64 threads (2 waves = 2 M-tiles).  Block owns 16 columns of
// each of the 4 gates; K = 1024 (inp@Wih^T) then 768 (h@Whh^T), fused cell.
// K loops kept rolled for the same spill reason as lstm_scan.
__global__ __launch_bounds__(64) void dec_cell(
    const __bf16* __restrict__ inp,    // 32 x 1024
    const __bf16* __restrict__ hb,     // 32 x 768 bf16 (cur)
    const __bf16* __restrict__ Wih,    // 3072 x 1024
    const __bf16* __restrict__ Whh,    // 3072 x 768
    const float* __restrict__ bias,    // 3072
    float* __restrict__ h_out,         // 32 x 768 f32 (next)
    float* __restrict__ c,             // 32 x 768 f32 (in-place)
    __bf16* __restrict__ hb_out,       // 32 x 768 bf16 (next)
    __bf16* __restrict__ hctx) {       // 32 x 1536 (write [0:768))
  int mt = threadIdx.x >> 5;           // 0..1
  int lane = threadIdx.x & 31;
  int n0 = blockIdx.x * 16;
  v8f acc[4] = {};
  const int DI = E + HD;               // 1024
#pragma clang loop unroll(disable)
  for (int k0 = 0; k0 < DI; k0 += 32) {
    v16bf a = load_frag(inp, DI, mt * 16, k0, Bn - 1);
#pragma unroll
    for (int g = 0; g < 4; ++g) {
      v16bf bfr = load_frag(Wih, DI, g * HD + n0, k0, GD - 1);
      acc[g] = wmma_bf16(a, bfr, acc[g]);
    }
  }
#pragma clang loop unroll(disable)
  for (int k0 = 0; k0 < HD; k0 += 32) {
    v16bf a = load_frag(hb, HD, mt * 16, k0, Bn - 1);
#pragma unroll
    for (int g = 0; g < 4; ++g) {
      v16bf bfr = load_frag(Whh, HD, g * HD + n0, k0, GD - 1);
      acc[g] = wmma_bf16(a, bfr, acc[g]);
    }
  }
  int col = n0 + (lane & 15);
  int rb = mt * 16 + (lane >> 4) * 8;
#pragma unroll
  for (int r = 0; r < 8; ++r) {
    int b = rb + r;
    float gi = acc[0][r] + bias[0 * HD + col];
    float gf = acc[1][r] + bias[1 * HD + col];
    float gg = acc[2][r] + bias[2 * HD + col];
    float go = acc[3][r] + bias[3 * HD + col];
    float cn = sigm(gf) * c[b * HD + col] + sigm(gi) * tanhf(gg);
    float hn = sigm(go) * tanhf(cn);
    c[b * HD + col] = cn;
    h_out[b * HD + col] = hn;
    hb_out[b * HD + col] = (__bf16)hn;
    hctx[(size_t)b * (2 * HD) + col] = (__bf16)hn;
  }
}

} // namespace

extern "C" void kernel_launch(void* const* d_in, const int* in_sizes, int n_in,
                              void* d_out, int out_size, void* d_ws, size_t ws_size,
                              hipStream_t stream) {
  (void)in_sizes; (void)n_in; (void)out_size; (void)ws_size;
  const int*   src     = (const int*)d_in[0];
  const int*   trg     = (const int*)d_in[1];
  const float* emb     = (const float*)d_in[2];
  const float* W1f_ih  = (const float*)d_in[3];
  const float* W1f_hh  = (const float*)d_in[4];
  const float* b1f     = (const float*)d_in[5];
  const float* W1b_ih  = (const float*)d_in[6];
  const float* W1b_hh  = (const float*)d_in[7];
  const float* b1b     = (const float*)d_in[8];
  const float* W2f_ih  = (const float*)d_in[9];
  const float* W2f_hh  = (const float*)d_in[10];
  const float* b2f     = (const float*)d_in[11];
  const float* W2b_ih  = (const float*)d_in[12];
  const float* W2b_hh  = (const float*)d_in[13];
  const float* b2b     = (const float*)d_in[14];
  const float* dec_Wih = (const float*)d_in[15];
  const float* dec_Whh = (const float*)d_in[16];
  const float* dec_b   = (const float*)d_in[17];
  const float* attn_Wh = (const float*)d_in[18];
  const float* attn_We = (const float*)d_in[19];
  const float* attn_b  = (const float*)d_in[20];
  const float* vvec    = (const float*)d_in[21];
  const float* fc_W    = (const float*)d_in[22];
  const float* fc_b    = (const float*)d_in[23];
  float* out = (float*)d_out;

  char* ws = (char*)d_ws;
  size_t off = 0;
  auto alloc = [&](size_t bytes) -> void* {
    void* p = ws + off;
    off = (off + bytes + 255) & ~(size_t)255;
    return p;
  };
  // bf16 weights (fc_W bf16 = 98 MB -> decoder working set stays in 192MB L2)
  __bf16* fcW_b  = (__bf16*)alloc((size_t)Vv * G1 * 2);
  __bf16* W1fi_b = (__bf16*)alloc((size_t)G1 * E * 2);
  __bf16* W1bi_b = (__bf16*)alloc((size_t)G1 * E * 2);
  __bf16* W1fh_b = (__bf16*)alloc((size_t)G1 * H * 2);
  __bf16* W1bh_b = (__bf16*)alloc((size_t)G1 * H * 2);
  __bf16* W2fi_b = (__bf16*)alloc((size_t)G1 * HD * 2);
  __bf16* W2bi_b = (__bf16*)alloc((size_t)G1 * HD * 2);
  __bf16* W2fh_b = (__bf16*)alloc((size_t)G1 * H * 2);
  __bf16* W2bh_b = (__bf16*)alloc((size_t)G1 * H * 2);
  __bf16* dWi_b  = (__bf16*)alloc((size_t)GD * (E + HD) * 2);
  __bf16* dWh_b  = (__bf16*)alloc((size_t)GD * HD * 2);
  __bf16* aWe_b  = (__bf16*)alloc((size_t)H * HD * 2);
  // activations
  __bf16* x_b    = (__bf16*)alloc((size_t)Bn * S * E * 2);
  float*  xgA    = (float*)alloc((size_t)Bn * S * G1 * 4);
  float*  xgB    = (float*)alloc((size_t)Bn * S * G1 * 4);
  __bf16* x1_b   = (__bf16*)alloc((size_t)Bn * S * HD * 2);
  __bf16* enc_b  = (__bf16*)alloc((size_t)Bn * S * HD * 2);
  float*  enc_f  = (float*)alloc((size_t)Bn * S * HD * 4);
  float*  encp   = (float*)alloc((size_t)Bn * S * H * 4);
  float*  hf0    = (float*)alloc((size_t)Bn * HD * 4);
  float*  hf1    = (float*)alloc((size_t)Bn * HD * 4);
  __bf16* hb0    = (__bf16*)alloc((size_t)Bn * HD * 2);
  __bf16* hb1    = (__bf16*)alloc((size_t)Bn * HD * 2);
  float*  dec_c  = (float*)alloc((size_t)Bn * HD * 4);
  __bf16* inp_b  = (__bf16*)alloc((size_t)Bn * (E + HD) * 2);
  __bf16* hctx_b = (__bf16*)alloc((size_t)Bn * (2 * HD) * 2);

  auto cast = [&](const float* s, __bf16* d, size_t n) {
    size_t g = (n + 255) / 256; if (g > 4096) g = 4096;
    cast_f32_bf16<<<dim3((unsigned)g), dim3(256), 0, stream>>>(s, d, n);
  };
  cast(fc_W,    fcW_b,  (size_t)Vv * G1);
  cast(W1f_ih,  W1fi_b, (size_t)G1 * E);
  cast(W1b_ih,  W1bi_b, (size_t)G1 * E);
  cast(W1f_hh,  W1fh_b, (size_t)G1 * H);
  cast(W1b_hh,  W1bh_b, (size_t)G1 * H);
  cast(W2f_ih,  W2fi_b, (size_t)G1 * HD);
  cast(W2b_ih,  W2bi_b, (size_t)G1 * HD);
  cast(W2f_hh,  W2fh_b, (size_t)G1 * H);
  cast(W2b_hh,  W2bh_b, (size_t)G1 * H);
  cast(dec_Wih, dWi_b,  (size_t)GD * (E + HD));
  cast(dec_Whh, dWh_b,  (size_t)GD * HD);
  cast(attn_We, aWe_b,  (size_t)H * HD);

  gather_src<<<dim3(Bn * S), dim3(E), 0, stream>>>(src, emb, x_b);

  const int MR = Bn * S;  // 8192
  // layer-1 input gates (both directions)
  gemm_bf16<4, 2><<<dim3(G1 / 64, MR / 64), dim3(256), 0, stream>>>(
      x_b, E, W1fi_b, E, b1f, xgA, G1, MR, G1, E);
  gemm_bf16<4, 2><<<dim3(G1 / 64, MR / 64), dim3(256), 0, stream>>>(
      x_b, E, W1bi_b, E, b1b, xgB, G1, MR, G1, E);
  lstm_scan<<<dim3(2), dim3(768), 0, stream>>>(
      xgA, xgB, W1fh_b, W1bh_b, x1_b, HD, nullptr, nullptr, nullptr);
  // layer-2 input gates (xg buffers reused)
  gemm_bf16<4, 2><<<dim3(G1 / 64, MR / 64), dim3(256), 0, stream>>>(
      x1_b, HD, W2fi_b, HD, b2f, xgA, G1, MR, G1, HD);
  gemm_bf16<4, 2><<<dim3(G1 / 64, MR / 64), dim3(256), 0, stream>>>(
      x1_b, HD, W2bi_b, HD, b2b, xgB, G1, MR, G1, HD);
  lstm_scan<<<dim3(2), dim3(768), 0, stream>>>(
      xgA, xgB, W2fh_b, W2bh_b, enc_b, HD, enc_f, hf0, dec_c);
  // enc_part = enc_out @ attn_We^T + attn_b
  gemm_bf16<4, 2><<<dim3(H / 64, MR / 64), dim3(256), 0, stream>>>(
      enc_b, HD, aWe_b, HD, attn_b, encp, H, MR, H, HD);
  cast(hf0, hb0, (size_t)Bn * HD);

  float*  hf[2] = {hf0, hf1};
  __bf16* hb[2] = {hb0, hb1};
  for (int t = 0; t < T; ++t) {
    int cur = t & 1, nxt = cur ^ 1;
    attn_step<<<dim3(Bn), dim3(256), 0, stream>>>(
        hf[cur], encp, enc_f, attn_Wh, vvec, trg, emb, inp_b, hctx_b, t);
    dec_cell<<<dim3(HD / 16), dim3(64), 0, stream>>>(
        inp_b, hb[cur], dWi_b, dWh_b, dec_b, hf[nxt], dec_c, hb[nxt], hctx_b);
    // FC: M=32 -> 2Mx4N wave layout, 32x128 block tile, no idle waves
    gemm_bf16<2, 4><<<dim3(Vv / 128, 1), dim3(256), 0, stream>>>(
        hctx_b, 2 * HD, fcW_b, G1, fc_b,
        out + (size_t)t * Vv, (size_t)T * Vv, Bn, Vv, G1);
  }
}
